// TestingGraphTransformersAugAttention_1228360647046
// MI455X (gfx1250) — compile-verified, matrix operand order checked by reading, not run on hardware
//
#include <hip/hip_runtime.h>

// ---------------------------------------------------------------------------
// Graph transformer w/ augmented attention, MI455X (gfx1250, wave32, WMMA).
// All heavy matmuls run through one bf16->f32 WMMA GEMM (v_wmma_f32_16x16x32_bf16),
// double-buffered LDS, fused epilogues (alpha / row-scale / bias / add / relu).
// ---------------------------------------------------------------------------

#define N_NODES 4096
#define D_IN    4096
#define D_MODELX 1024
#define K_W     10

typedef __attribute__((ext_vector_type(16))) __bf16 bf16x16;
typedef __attribute__((ext_vector_type(8)))  __bf16 bf16x8;
typedef __attribute__((ext_vector_type(4)))  __bf16 bf16x4;
typedef __attribute__((ext_vector_type(8)))  float  f32x8;

#define BM 128
#define BN 128
#define BK 32
#define SK 48   // padded LDS row stride (bf16 elems): 96B -> 16B-aligned b128 frag loads

// C[i,j] = act( alpha*acc*(rowscale?rs[i*rs_stride]:1) + bias[j] + addmat[i,j] + (accumulate?C:0) )
template<bool TRANS_B>
__global__ __launch_bounds__(256)
void gemm_bf16_wmma(const float* __restrict__ A, const float* __restrict__ B,
                    float* __restrict__ C, int M, int N, int K,
                    int lda, int ldb, float alpha,
                    const float* __restrict__ bias,
                    const float* __restrict__ addmat,
                    const float* __restrict__ rowscale, int rs_stride,
                    int accumulate, int relu)
{
  __shared__ __align__(16) __bf16 As[2][BM * SK];
  __shared__ __align__(16) __bf16 Bs[2][BN * SK];

  const int tid     = threadIdx.x;
  const int rowBase = blockIdx.y * BM;
  const int colBase = blockIdx.x * BN;

  const int wave  = tid >> 5;     // 8 waves: 2 (M) x 4 (N)
  const int lane  = tid & 31;
  const int wm    = wave & 1;     // 64 rows per wave
  const int wn    = wave >> 1;    // 32 cols per wave
  const int lmn   = lane & 15;    // M (A frag) / N (B frag) index
  const int khalf = lane >> 4;    // ISA: lanes 16-31 hold the upper K-halves
  const int ko    = khalf * 8;

  f32x8 acc[4][2];
  const f32x8 fzero = {};
#pragma unroll
  for (int i = 0; i < 4; ++i)
#pragma unroll
    for (int j = 0; j < 2; ++j) acc[i][j] = fzero;

  float4 ra[4], rb[4];  // staging regs (global -> reg -> bf16 LDS pipeline)

  auto loadA = [&](int kk) {
#pragma unroll
    for (int p = 0; p < 4; ++p) {
      int idx = p * 256 + tid;
      int r = idx >> 3, cg = idx & 7;               // 128 rows x 8 float4 groups
      ra[p] = *(const float4*)(A + (size_t)(rowBase + r) * lda + kk + cg * 4);
    }
  };
  auto loadB = [&](int kk) {
#pragma unroll
    for (int p = 0; p < 4; ++p) {
      int idx = p * 256 + tid;
      if (TRANS_B) {                                 // B stored [N,K]: row reads
        int n = idx >> 3, cg = idx & 7;
        rb[p] = *(const float4*)(B + (size_t)(colBase + n) * ldb + kk + cg * 4);
      } else {                                       // B stored [K,N]: coalesced along N
        int k = idx >> 5, ng = idx & 31;
        rb[p] = *(const float4*)(B + (size_t)(kk + k) * ldb + colBase + ng * 4);
      }
    }
  };
  auto storeA = [&](int buf) {
#pragma unroll
    for (int p = 0; p < 4; ++p) {
      int idx = p * 256 + tid;
      int r = idx >> 3, cg = idx & 7;
      bf16x4 t = {(__bf16)ra[p].x, (__bf16)ra[p].y, (__bf16)ra[p].z, (__bf16)ra[p].w};
      *(bf16x4*)(&As[buf][r * SK + cg * 4]) = t;     // 8B-aligned ds_store_b64
    }
  };
  auto storeB = [&](int buf) {
#pragma unroll
    for (int p = 0; p < 4; ++p) {
      int idx = p * 256 + tid;
      if (TRANS_B) {
        int n = idx >> 3, cg = idx & 7;
        bf16x4 t = {(__bf16)rb[p].x, (__bf16)rb[p].y, (__bf16)rb[p].z, (__bf16)rb[p].w};
        *(bf16x4*)(&Bs[buf][n * SK + cg * 4]) = t;
      } else {                                       // transpose [K,N] -> LDS [n][k]
        int k = idx >> 5, ng = idx & 31;
        Bs[buf][(ng * 4 + 0) * SK + k] = (__bf16)rb[p].x;
        Bs[buf][(ng * 4 + 1) * SK + k] = (__bf16)rb[p].y;
        Bs[buf][(ng * 4 + 2) * SK + k] = (__bf16)rb[p].z;
        Bs[buf][(ng * 4 + 3) * SK + k] = (__bf16)rb[p].w;
      }
    }
  };

  const int nsteps = K / BK;
  loadA(0); loadB(0);

  for (int s = 0; s < nsteps; ++s) {
    const int buf = s & 1;
    storeA(buf); storeB(buf);
    __syncthreads();                                  // single barrier per K-step (double buffer)

    if (s + 1 < nsteps) { loadA((s + 1) * BK); loadB((s + 1) * BK); }
    if (s + 2 < nsteps) {                             // gfx1250 global_prefetch_b8
      __builtin_prefetch(A + (size_t)(rowBase + (tid >> 3)) * lda + (s + 2) * BK, 0, 1);
      __builtin_prefetch(B + (size_t)(TRANS_B ? (colBase + (tid >> 3)) : ((s + 2) * BK + (tid >> 5)))
                               * ldb + (TRANS_B ? (s + 2) * BK : colBase), 0, 1);
    }

    bf16x16 afrag[4], bfrag[2];
#pragma unroll
    for (int fm = 0; fm < 4; ++fm) {
      const __bf16* p = &As[buf][(wm * 64 + fm * 16 + lmn) * SK + ko];
      bf16x8 lo = *(const bf16x8*)(p);                // K = ko..ko+7
      bf16x8 hi = *(const bf16x8*)(p + 16);           // K = ko+16..ko+23
      afrag[fm] = __builtin_shufflevector(lo, hi, 0,1,2,3,4,5,6,7,8,9,10,11,12,13,14,15);
    }
#pragma unroll
    for (int fn = 0; fn < 2; ++fn) {
      const __bf16* p = &Bs[buf][(wn * 32 + fn * 16 + lmn) * SK + ko];
      bf16x8 lo = *(const bf16x8*)(p);
      bf16x8 hi = *(const bf16x8*)(p + 16);
      bfrag[fn] = __builtin_shufflevector(lo, hi, 0,1,2,3,4,5,6,7,8,9,10,11,12,13,14,15);
    }
#pragma unroll
    for (int fm = 0; fm < 4; ++fm)
#pragma unroll
      for (int fn = 0; fn < 2; ++fn)
        acc[fm][fn] = __builtin_amdgcn_wmma_f32_16x16x32_bf16(
            false, afrag[fm], false, bfrag[fn], (short)0, acc[fm][fn], false, false);
  }

  // Epilogue: C/D layout — lane (l&15)=N, VGPR r -> M = (l>>4)*8 + r
#pragma unroll
  for (int fm = 0; fm < 4; ++fm)
#pragma unroll
    for (int fn = 0; fn < 2; ++fn)
#pragma unroll
      for (int r = 0; r < 8; ++r) {
        int i = rowBase + wm * 64 + fm * 16 + khalf * 8 + r;
        int j = colBase + wn * 32 + fn * 16 + lmn;
        float v = alpha * acc[fm][fn][r];
        if (rowscale) v *= rowscale[(size_t)i * rs_stride];
        if (bias)     v += bias[j];
        if (addmat)   v += addmat[(size_t)i * N + j];
        size_t off = (size_t)i * N + j;
        if (accumulate) v += C[off];
        if (relu) v = v > 0.f ? v : 0.f;
        C[off] = v;
      }
}

// ---------------- auxiliary kernels ----------------

__global__ __launch_bounds__(256) void rowsum_kernel(const float* __restrict__ A,
                                                     float* __restrict__ rs, int n) {
  __shared__ float red[256];
  int i = blockIdx.x;
  float s = 0.f;
  for (int j = threadIdx.x; j < n; j += 256) s += A[(size_t)i * n + j];
  red[threadIdx.x] = s; __syncthreads();
  for (int st = 128; st > 0; st >>= 1) {
    if (threadIdx.x < st) red[threadIdx.x] += red[threadIdx.x + st];
    __syncthreads();
  }
  if (threadIdx.x == 0) rs[i] = red[0];
}

__global__ __launch_bounds__(256) void colsum_kernel(const float* __restrict__ A,
                                                     float* __restrict__ cs, int n) {
  int j = blockIdx.x * 256 + threadIdx.x;
  if (j >= n) return;
  float s = 0.f;
  for (int i = 0; i < n; ++i) s += A[(size_t)i * n + j];
  cs[j] = s;
}

// S(=inter) -> inter/(row+col-inter+1e-6) + graph_metrics   (in place)
__global__ __launch_bounds__(256) void neighborhood_sim_kernel(
    float* __restrict__ S, const float* __restrict__ gm,
    const float* __restrict__ rs, const float* __restrict__ cs, int n) {
  size_t total = (size_t)n * n;
  for (size_t idx = (size_t)blockIdx.x * 256 + threadIdx.x; idx < total;
       idx += (size_t)gridDim.x * 256) {
    int i = (int)(idx / n), j = (int)(idx % n);
    float inter = S[idx];
    S[idx] = inter / (rs[i] + cs[j] - inter + 1e-6f) + gm[idx];
  }
}

// h = LN(wx + weights@b_stack) * gamma + beta ; one block per node
__global__ __launch_bounds__(256) void combine_ln_kernel(
    const float* __restrict__ wx, const float* __restrict__ wts,
    const float* __restrict__ b_stack, const float* __restrict__ gamma,
    const float* __restrict__ beta, float* __restrict__ h) {
  __shared__ float red[256];
  const int nrow = blockIdx.x, tid = threadIdx.x;
  float v[4], s = 0.f;
#pragma unroll
  for (int q = 0; q < 4; ++q) {
    int j = tid + q * 256;
    float wb = 0.f;
#pragma unroll
    for (int k = 0; k < K_W; ++k) wb += wts[nrow * K_W + k] * b_stack[k * D_MODELX + j];
    v[q] = wx[(size_t)nrow * D_MODELX + j] + wb;
    s += v[q];
  }
  red[tid] = s; __syncthreads();
  for (int st = 128; st > 0; st >>= 1) { if (tid < st) red[tid] += red[tid + st]; __syncthreads(); }
  float mu = red[0] * (1.f / D_MODELX);
  __syncthreads();
  float sq = 0.f;
#pragma unroll
  for (int q = 0; q < 4; ++q) { float d = v[q] - mu; sq += d * d; }
  red[tid] = sq; __syncthreads();
  for (int st = 128; st > 0; st >>= 1) { if (tid < st) red[tid] += red[tid + st]; __syncthreads(); }
  float rstd = rsqrtf(red[0] * (1.f / D_MODELX) + 1e-5f);
#pragma unroll
  for (int q = 0; q < 4; ++q) {
    int j = tid + q * 256;
    h[(size_t)nrow * D_MODELX + j] = (v[q] - mu) * rstd * gamma[j] + beta[j];
  }
}

// in-place row softmax over 4096 columns; one block per row
__global__ __launch_bounds__(256) void softmax_rows_kernel(float* __restrict__ S) {
  __shared__ float red[256];
  float* p = S + (size_t)blockIdx.x * N_NODES;
  const int tid = threadIdx.x;
  float v[16], m = -3.4e38f;
#pragma unroll
  for (int q = 0; q < 16; ++q) { v[q] = p[tid + q * 256]; m = fmaxf(m, v[q]); }
  red[tid] = m; __syncthreads();
  for (int st = 128; st > 0; st >>= 1) { if (tid < st) red[tid] = fmaxf(red[tid], red[tid + st]); __syncthreads(); }
  m = red[0]; __syncthreads();
  float s = 0.f;
#pragma unroll
  for (int q = 0; q < 16; ++q) { v[q] = __expf(v[q] - m); s += v[q]; }
  red[tid] = s; __syncthreads();
  for (int st = 128; st > 0; st >>= 1) { if (tid < st) red[tid] += red[tid + st]; __syncthreads(); }
  float inv = 1.f / red[0];
#pragma unroll
  for (int q = 0; q < 16; ++q) p[tid + q * 256] = v[q] * inv;
}

// ---------------- launch ----------------

extern "C" void kernel_launch(void* const* d_in, const int* in_sizes, int n_in,
                              void* d_out, int out_size, void* d_ws, size_t ws_size,
                              hipStream_t stream) {
  (void)in_sizes; (void)n_in; (void)out_size; (void)ws_size;
  const float* x     = (const float*)d_in[0];
  const float* adj   = (const float*)d_in[1];
  const float* gm    = (const float*)d_in[2];
  const float* wts   = (const float*)d_in[3];
  const float* Wst   = (const float*)d_in[4];
  const float* bst   = (const float*)d_in[5];
  const float* Wq    = (const float*)d_in[6];
  const float* bq    = (const float*)d_in[7];
  const float* Wk    = (const float*)d_in[8];
  const float* bk    = (const float*)d_in[9];
  const float* Wv    = (const float*)d_in[10];
  const float* bv    = (const float*)d_in[11];
  const float* Wo    = (const float*)d_in[12];
  const float* bo    = (const float*)d_in[13];
  const float* gamma = (const float*)d_in[14];
  const float* beta  = (const float*)d_in[15];
  float* outp = (float*)d_out;

  const size_t NN = (size_t)N_NODES * N_NODES;
  const size_t ND = (size_t)N_NODES * D_MODELX;
  float* ws     = (float*)d_ws;           // ~288 MB scratch layout
  float* dng    = ws;                     // [N,N]
  float* direct = dng + NN;               // [N,N]
  float* Sbuf   = direct + NN;            // [N,N]: inter -> sim+gm -> scores -> attn
  float* wx     = Sbuf + NN;              // [N,Dm]
  float* h      = wx + ND;                // [N,Dm]
  float* Q      = h + ND;                 // [N,Dm]
  float* Km     = Q + ND;                 // [N,Dm]
  float* V      = Km + ND;                // [N,Dm]
  float* outm   = V + ND;                 // [N,Dm]
  float* rsum   = outm + ND;              // [N]
  float* csum   = rsum + N_NODES;         // [N]

  dim3 blk(256);
  dim3 gBig(N_NODES / BN, N_NODES / BM);  // 32 x 32
  dim3 gMed(D_MODELX / BN, N_NODES / BM); // 8 x 32

  rowsum_kernel<<<N_NODES, blk, 0, stream>>>(adj, rsum, N_NODES);
  colsum_kernel<<<N_NODES / 256, blk, 0, stream>>>(adj, csum, N_NODES);

  // direct = A @ x
  gemm_bf16_wmma<false><<<gBig, blk, 0, stream>>>(adj, x, direct,
      N_NODES, D_IN, N_NODES, N_NODES, D_IN, 1.f, nullptr, nullptr, nullptr, 0, 0, 0);
  // inter = A @ A
  gemm_bf16_wmma<false><<<gBig, blk, 0, stream>>>(adj, adj, Sbuf,
      N_NODES, N_NODES, N_NODES, N_NODES, N_NODES, 1.f, nullptr, nullptr, nullptr, 0, 0, 0);
  // Sbuf = inter/(row+col-inter+eps) + graph_metrics
  neighborhood_sim_kernel<<<4096, blk, 0, stream>>>(Sbuf, gm, rsum, csum, N_NODES);
  // dng = Sbuf @ x + direct
  gemm_bf16_wmma<false><<<gBig, blk, 0, stream>>>(Sbuf, x, dng,
      N_NODES, D_IN, N_NODES, N_NODES, D_IN, 1.f, nullptr, direct, nullptr, 0, 0, 0);
  // wx = sum_k weights[:,k] * (x @ W_stack[k])   (accumulate, per-row scale epilogue)
  for (int k = 0; k < K_W; ++k)
    gemm_bf16_wmma<false><<<gMed, blk, 0, stream>>>(x, Wst + (size_t)k * D_IN * D_MODELX, wx,
        N_NODES, D_MODELX, D_IN, D_IN, D_MODELX, 1.f, nullptr, nullptr,
        wts + k, K_W, k > 0 ? 1 : 0, 0);
  // h = LN(wx + weights@b_stack)*gamma + beta
  combine_ln_kernel<<<N_NODES, blk, 0, stream>>>(wx, wts, bst, gamma, beta, h);
  // Q,K,V
  gemm_bf16_wmma<false><<<gMed, blk, 0, stream>>>(h, Wq, Q,
      N_NODES, D_MODELX, D_MODELX, D_MODELX, D_MODELX, 1.f, bq, nullptr, nullptr, 0, 0, 0);
  gemm_bf16_wmma<false><<<gMed, blk, 0, stream>>>(h, Wk, Km,
      N_NODES, D_MODELX, D_MODELX, D_MODELX, D_MODELX, 1.f, bk, nullptr, nullptr, 0, 0, 0);
  gemm_bf16_wmma<false><<<gMed, blk, 0, stream>>>(h, Wv, V,
      N_NODES, D_MODELX, D_MODELX, D_MODELX, D_MODELX, 1.f, bv, nullptr, nullptr, 0, 0, 0);
  // scores = Q @ K^T / 32 + dng   (NT GEMM, into Sbuf)
  gemm_bf16_wmma<true><<<gBig, blk, 0, stream>>>(Q, Km, Sbuf,
      N_NODES, N_NODES, D_MODELX, D_MODELX, D_MODELX, 1.f / 32.f, nullptr, dng,
      nullptr, 0, 0, 0);
  // softmax rows in place
  softmax_rows_kernel<<<N_NODES, blk, 0, stream>>>(Sbuf);
  // outm = attn @ V
  gemm_bf16_wmma<false><<<gMed, blk, 0, stream>>>(Sbuf, V, outm,
      N_NODES, D_MODELX, N_NODES, N_NODES, D_MODELX, 1.f, nullptr, nullptr, nullptr, 0, 0, 0);
  // out = relu(outm @ Wo + bo + dng)
  gemm_bf16_wmma<false><<<gBig, blk, 0, stream>>>(outm, Wo, outp,
      N_NODES, D_IN, D_MODELX, D_MODELX, D_IN, 1.f, bo, dng, nullptr, 0, 0, 1);
}
